// EdgeGNNGRU_81192061764392
// MI455X (gfx1250) — compile-verified, matrix operand order checked by reading, not run on hardware
//
#include <hip/hip_runtime.h>
#include <hip/hip_bf16.h>
#include <math.h>

// ---------------------------------------------------------------------------
// Problem constants (match the JAX reference)
// ---------------------------------------------------------------------------
#define GB    64          // graphs per batch
#define NPG   512         // nodes per graph
#define EPG   1024        // edges per graph
#define NN    (GB * NPG)  // 32768 nodes
#define NE    (GB * EPG)  // 65536 edges
#define DH    128         // feature dim (D_IN == D_H == 128)

typedef _Float16 half_t;
typedef __attribute__((ext_vector_type(16))) _Float16 v16h;
typedef __attribute__((ext_vector_type(8)))  float    v8f;
typedef __attribute__((ext_vector_type(4)))  int      v4i_t;

// Async LDS staging path (gfx1250 GLOBAL_LOAD_ASYNC_TO_LDS, ASYNCcnt-tracked).
#if defined(__gfx1250__) || defined(__HIP_DEVICE_COMPILE__)
#if __has_builtin(__builtin_amdgcn_global_load_async_to_lds_b128) && \
    __has_builtin(__builtin_amdgcn_s_wait_asynccnt)
#define USE_ASYNC_LDS 1
#endif
#endif
#ifndef USE_ASYNC_LDS
#define USE_ASYNC_LDS 0
#endif

#if USE_ASYNC_LDS
// Builtin signature (from hipcc diagnostics): param0 = v4i addrspace(1)*,
// param1 = v4i addrspace(3)*, then imm offset, imm cpol.
#define ASYNC_LDS_B128(gptr, lptr)                                             \
    __builtin_amdgcn_global_load_async_to_lds_b128(                            \
        (__attribute__((address_space(1))) v4i_t*)(gptr),                      \
        (__attribute__((address_space(3))) v4i_t*)(lptr), 0, 0)
#endif

// ---------------------------------------------------------------------------
// Small helper kernels
// ---------------------------------------------------------------------------

// Convert W (f32, [k][n]) to f16 TRANSPOSED ([n][k]) so GEMM B-fragments are
// contiguous along K and LDS staging is a straight block copy.
__global__ void k_cvt_w_f16_t(const float* __restrict__ w, half_t* __restrict__ wt) {
    int i = blockIdx.x * blockDim.x + threadIdx.x;   // over DH*DH
    if (i >= DH * DH) return;
    int k = i >> 7, n = i & (DH - 1);
    wt[(size_t)n * DH + k] = (half_t)w[i];
}

__global__ void k_gather_emb(const int* __restrict__ idx,
                             const float* __restrict__ emb,
                             half_t* __restrict__ h16) {
    int i = blockIdx.x * blockDim.x + threadIdx.x;   // over NN*DH
    if (i >= NN * DH) return;
    int n = i >> 7, c = i & (DH - 1);
    h16[i] = (half_t)emb[(size_t)idx[n] * DH + c];
}

__global__ void k_deg_init(float* __restrict__ deg) {
    int n = blockIdx.x * blockDim.x + threadIdx.x;
    if (n < NN) deg[n] = 1.0f;                        // self loop
}

__global__ void k_deg_count(const int* __restrict__ dst, float* __restrict__ deg) {
    int e = blockIdx.x * blockDim.x + threadIdx.x;
    if (e < NE) atomicAdd(&deg[dst[e]], 1.0f);
}

__global__ void k_rsqrt_inplace(float* __restrict__ deg) {
    int n = blockIdx.x * blockDim.x + threadIdx.x;
    if (n < NN) deg[n] = rsqrtf(deg[n]);              // deg -> deg^-1/2
}

__global__ void k_zero_f32(float* __restrict__ p, int n) {
    int i = blockIdx.x * blockDim.x + threadIdx.x;
    if (i < n) p[i] = 0.0f;
}

// ---------------------------------------------------------------------------
// WMMA GEMM: C[M,128] = A[M,128] (f16) @ W (f16, pre-transposed [n][k]), C f32
// block = 128 threads (4 waves); block computes a 64-row strip.
// ---------------------------------------------------------------------------
__global__ __launch_bounds__(128, 1)
void k_gemm_wmma(const half_t* __restrict__ A,
                 const half_t* __restrict__ Wt_g,
                 float* __restrict__ C) {
    __shared__ __align__(16) half_t As[64 * DH];      // 16 KB
    __shared__ __align__(16) half_t Wt[DH * DH];      // 32 KB, [n*128+k]

    const int tid = threadIdx.x;
    const int m0  = blockIdx.x * 64;

#if USE_ASYNC_LDS
    // Stage Wt (32 KB) and the 64x128 A strip (16 KB) via async DMA to LDS.
    for (int i = tid * 8; i < DH * DH; i += 128 * 8) {
        ASYNC_LDS_B128(Wt_g + i, &Wt[i]);
    }
    for (int i = tid * 8; i < 64 * DH; i += 128 * 8) {
        ASYNC_LDS_B128(A + (size_t)m0 * DH + i, &As[i]);
    }
    __builtin_amdgcn_s_wait_asynccnt(0);   // drain this wave's ASYNCcnt
#else
    for (int i = tid * 8; i < DH * DH; i += 128 * 8) {
        *(uint4*)&Wt[i] = *(const uint4*)&Wt_g[i];
    }
    for (int i = tid * 8; i < 64 * DH; i += 128 * 8) {
        *(uint4*)&As[i] = *(const uint4*)&A[(size_t)m0 * DH + i];
    }
#endif
    // Speculative prefetch of the next block's A strip -> global_prefetch_b8.
    if (m0 + 64 < NN) __builtin_prefetch(&A[(size_t)(m0 + 64) * DH], 0, 0);
    __syncthreads();                       // publish LDS across waves

    const int wave = tid >> 5;
    const int lane = tid & 31;
    const int ln   = lane & 15;
    const int hi   = lane >> 4;
    const int mloc = wave * 16 + ln;                  // A row handled by this lane

    // Preload A fragments for all 4 K-tiles (ISA 16-bit A 16x32 layout:
    // lane ln holds row M=ln; element i -> K = (i<8?0:16) + hi*8 + (i&7)).
    v16h afr[4];
#pragma unroll
    for (int kt = 0; kt < 4; ++kt) {
        const int kb = kt * 32 + hi * 8;
        *((uint4*)&afr[kt])     = *(const uint4*)&As[mloc * DH + kb];
        *((uint4*)&afr[kt] + 1) = *(const uint4*)&As[mloc * DH + kb + 16];
    }

#pragma unroll
    for (int nt = 0; nt < 8; ++nt) {
        v8f acc = {};
        const int n = nt * 16 + ln;
#pragma unroll
        for (int kt = 0; kt < 4; ++kt) {
            v16h bfr;
            const int kb = kt * 32 + hi * 8;
            *((uint4*)&bfr)     = *(const uint4*)&Wt[n * DH + kb];
            *((uint4*)&bfr + 1) = *(const uint4*)&Wt[n * DH + kb + 16];
            acc = __builtin_amdgcn_wmma_f32_16x16x32_f16(
                false, afr[kt], false, bfr, (short)0, acc, false, false);
        }
        // C/D layout: VGPR j -> M = j + hi*8; N = ln.
        const int mg = m0 + wave * 16 + hi * 8;
        float* cp = C + (size_t)mg * DH + nt * 16 + ln;
#pragma unroll
        for (int j = 0; j < 8; ++j) cp[(size_t)j * DH] = acc[j];
    }
}

// ---------------------------------------------------------------------------
// GCN scatter: acc[dst] += dis[src]*dis[dst] * xw[src]   (edge term)
// ---------------------------------------------------------------------------
__global__ void k_scatter(const int* __restrict__ src, const int* __restrict__ dst,
                          const float* __restrict__ dis,
                          const float* __restrict__ xw,
                          float* __restrict__ acc) {
    int i = blockIdx.x * blockDim.x + threadIdx.x;    // over NE*DH
    if (i >= NE * DH) return;
    int e = i >> 7, c = i & (DH - 1);
    int s = src[e], d = dst[e];
    float norm = dis[s] * dis[d];
    atomicAdd(&acc[(size_t)d * DH + c], norm * xw[(size_t)s * DH + c]);
}

// finalize: h = gelu(acc + dis[n]^2 * xw[n] + b); write f16 (next GEMM) + f32
__global__ void k_finalize(const float* __restrict__ acc,
                           const float* __restrict__ xw,
                           const float* __restrict__ dis,
                           const float* __restrict__ b,
                           half_t* __restrict__ h16,
                           float* __restrict__ xf32) {
    int i = blockIdx.x * blockDim.x + threadIdx.x;    // over NN*DH
    if (i >= NN * DH) return;
    int n = i >> 7, c = i & (DH - 1);
    float dn = dis[n];
    float v  = acc[i] + dn * dn * xw[i] + b[c];
    float g  = 0.5f * v * (1.0f + erff(v * 0.70710678118654752f));   // exact GELU
    h16[i]  = (half_t)g;
    xf32[i] = g;
}

// ---------------------------------------------------------------------------
// Edge gates: gx[e,0:3] = (x[src]+x[dst]) . W_ih[g,:] + b_ih[g]
// ---------------------------------------------------------------------------
__global__ void k_edge_gates(const int* __restrict__ src, const int* __restrict__ dst,
                             const float* __restrict__ xf,
                             const float* __restrict__ W_ih,
                             const float* __restrict__ b_ih,
                             float* __restrict__ gx) {
    int e = blockIdx.x * blockDim.x + threadIdx.x;
    if (e >= NE) return;
    const float4* xs = (const float4*)(xf + (size_t)src[e] * DH);
    const float4* xd = (const float4*)(xf + (size_t)dst[e] * DH);
    const float4* w0 = (const float4*)(W_ih);
    const float4* w1 = (const float4*)(W_ih + DH);
    const float4* w2 = (const float4*)(W_ih + 2 * DH);
    float g0 = b_ih[0], g1 = b_ih[1], g2 = b_ih[2];
#pragma unroll 8
    for (int i = 0; i < DH / 4; ++i) {
        float4 a = xs[i], bq = xd[i];
        float ex = a.x + bq.x, ey = a.y + bq.y, ez = a.z + bq.z, ew = a.w + bq.w;
        float4 p;
        p = w0[i]; g0 += ex * p.x + ey * p.y + ez * p.z + ew * p.w;
        p = w1[i]; g1 += ex * p.x + ey * p.y + ez * p.z + ew * p.w;
        p = w2[i]; g2 += ex * p.x + ey * p.y + ez * p.z + ew * p.w;
    }
    float* o = gx + (size_t)e * 3;
    o[0] = g0; o[1] = g1; o[2] = g2;
}

// ---------------------------------------------------------------------------
// GRU (hidden size 1) scan over T=1024, one lane per graph; emit last-nonzero.
// ---------------------------------------------------------------------------
__global__ void k_gru(const float* __restrict__ gx,
                      const float* __restrict__ W_hh,
                      const float* __restrict__ b_hh,
                      const float* __restrict__ initial_hs,
                      float* __restrict__ out) {
    int b = threadIdx.x;
    if (b >= GB) return;
    float h  = initial_hs[0];
    float wr = W_hh[0], wz = W_hh[1], wn = W_hh[2];
    float br = b_hh[0], bz = b_hh[1], bn = b_hh[2];
    float sum = 0.0f, last = 0.0f, first = 0.0f;
    const float* g = gx + (size_t)b * EPG * 3;
    for (int t = 0; t < EPG; ++t, g += 3) {
        float r = 1.0f / (1.0f + expf(-(g[0] + wr * h + br)));
        float z = 1.0f / (1.0f + expf(-(g[1] + wz * h + bz)));
        float n = tanhf(g[2] + r * (wn * h + bn));
        h = (1.0f - z) * n + z * h;
        sum += h;
        if (h != 0.0f) last = h;
        if (t == 0) first = h;
    }
    out[b] = (sum > 0.0f) ? last : first;
}

// ---------------------------------------------------------------------------
// Host-side launch
// ---------------------------------------------------------------------------
extern "C" void kernel_launch(void* const* d_in, const int* in_sizes, int n_in,
                              void* d_out, int out_size, void* d_ws, size_t ws_size,
                              hipStream_t stream) {
    (void)in_sizes; (void)n_in; (void)out_size; (void)ws_size;

    const int*   x_idx   = (const int*)d_in[0];
    const int*   src     = (const int*)d_in[1];            // edge_index row 0
    const int*   dst     = ((const int*)d_in[1]) + NE;     // edge_index row 1
    // d_in[2] = batch (unused: edges are graph-grouped with exactly EPG each)
    const float* emb     = (const float*)d_in[3];
    const float* W1      = (const float*)d_in[4];
    const float* b1      = (const float*)d_in[5];
    const float* W2      = (const float*)d_in[6];
    const float* b2      = (const float*)d_in[7];
    const float* W_ih    = (const float*)d_in[8];
    const float* b_ih    = (const float*)d_in[9];
    const float* W_hh    = (const float*)d_in[10];
    const float* b_hh    = (const float*)d_in[11];
    const float* init_hs = (const float*)d_in[12];
    float*       out     = (float*)d_out;

    // ---- workspace carve-up -------------------------------------------------
    char* ws = (char*)d_ws;
    size_t off = 0;
    auto take = [&](size_t bytes) { char* p = ws + off; off += (bytes + 255) & ~(size_t)255; return p; };
    half_t* h16   = (half_t*)take((size_t)NN * DH * sizeof(half_t));   // GEMM input
    half_t* w16_1 = (half_t*)take((size_t)DH * DH * sizeof(half_t));   // W1^T f16
    half_t* w16_2 = (half_t*)take((size_t)DH * DH * sizeof(half_t));   // W2^T f16
    float*  xw    = (float*) take((size_t)NN * DH * sizeof(float));    // GEMM out
    float*  accb  = (float*) take((size_t)NN * DH * sizeof(float));    // scatter acc
    float*  xf32  = (float*) take((size_t)NN * DH * sizeof(float));    // activations
    float*  dis   = (float*) take((size_t)NN * sizeof(float));         // deg^-1/2
    float*  gx    = (float*) take((size_t)NE * 3 * sizeof(float));     // gate inputs

    const int T256 = 256;
    const int gNC  = (NN * DH + T256 - 1) / T256;   // node-feature grid
    const int gEC  = (NE * DH + T256 - 1) / T256;   // edge-feature grid
    const int gN   = (NN + T256 - 1) / T256;
    const int gE   = (NE + T256 - 1) / T256;

    // ---- prologue: weight f16+transpose, embedding gather, degrees ---------
    k_cvt_w_f16_t<<<(DH * DH + T256 - 1) / T256, T256, 0, stream>>>(W1, w16_1);
    k_cvt_w_f16_t<<<(DH * DH + T256 - 1) / T256, T256, 0, stream>>>(W2, w16_2);
    k_gather_emb <<<gNC, T256, 0, stream>>>(x_idx, emb, h16);
    k_deg_init   <<<gN,  T256, 0, stream>>>(dis);
    k_deg_count  <<<gE,  T256, 0, stream>>>(dst, dis);
    k_rsqrt_inplace<<<gN, T256, 0, stream>>>(dis);

    // ---- GCN layer 1 --------------------------------------------------------
    k_gemm_wmma <<<NN / 64, 128, 0, stream>>>(h16, w16_1, xw);
    k_zero_f32  <<<gNC, T256, 0, stream>>>(accb, NN * DH);
    k_scatter   <<<gEC, T256, 0, stream>>>(src, dst, dis, xw, accb);
    k_finalize  <<<gNC, T256, 0, stream>>>(accb, xw, dis, b1, h16, xf32);

    // ---- GCN layer 2 --------------------------------------------------------
    k_gemm_wmma <<<NN / 64, 128, 0, stream>>>(h16, w16_2, xw);
    k_zero_f32  <<<gNC, T256, 0, stream>>>(accb, NN * DH);
    k_scatter   <<<gEC, T256, 0, stream>>>(src, dst, dis, xw, accb);
    k_finalize  <<<gNC, T256, 0, stream>>>(accb, xw, dis, b2, h16, xf32);

    // ---- edge gates + GRU scan ---------------------------------------------
    k_edge_gates<<<gE, T256, 0, stream>>>(src, dst, xf32, W_ih, b_ih, gx);
    k_gru       <<<1, 64, 0, stream>>>(gx, W_hh, b_hh, init_hs, out);
}
